// Text2SemanticDecoder_86887188398299
// MI455X (gfx1250) — matrix operand order repaired and checked by reading, not compiled, not run on hardware
//
#include <hip/hip_runtime.h>
#include <math.h>

#define LAYERS 24
#define TCACHE 2048
#define DMODEL 512
#define NHEAD  16
#define DHEAD  32
#define VOCAB  1025
#define TYLEN  512
#define FFDIM  2048
#define TOPK   15
#define REPP   1.35f

typedef __attribute__((ext_vector_type(16))) _Float16 v16h;
typedef __attribute__((ext_vector_type(8)))  float    v8f;

// ---------------------------------------------------------------------------
// x = tok_emb[last_token] + alpha * sine_pe(row TY)
// ---------------------------------------------------------------------------
__global__ void prep_x_kernel(const int* __restrict__ y,
                              const float* __restrict__ tok_emb,
                              const float* __restrict__ alpha,
                              float* __restrict__ x)
{
    const int d   = threadIdx.x;                 // 512 threads
    const int tok = y[TYLEN - 1];
    const int j   = d >> 1;
    const float dv  = __expf((float)(2 * j) * (-9.21034037f / (float)DMODEL));
    const float ang = (float)TYLEN * dv;
    const float pe  = (d & 1) ? __cosf(ang) : __sinf(ang);
    x[d] = tok_emb[(size_t)tok * DMODEL + d] + alpha[0] * pe;
}

// ---------------------------------------------------------------------------
// GEMV via WMMA: out[0:N] = act(vec[0:K] @ W[K,N] + bias + resid)
// One wave per 16-column tile.  A-matrix carries vec in row M=0 (other rows
// zero), B-matrix is a 32x16 f16 tile of W, so row 0 of D is the GEMV result.
// Each weight element is read exactly once -> bandwidth optimal; the extra
// WMMA rows are free under the 23.3 TB/s memory roof.
// ---------------------------------------------------------------------------
__global__ void gemv_wmma_kernel(const float* __restrict__ vec,
                                 const float* __restrict__ W,
                                 const float* __restrict__ bias,
                                 const float* __restrict__ resid,
                                 float* __restrict__ out,
                                 int K, int N, int relu)
{
    const int lane = threadIdx.x & 31;
    const int wave = threadIdx.x >> 5;
    const int tile = blockIdx.x * (blockDim.x >> 5) + wave;
    const int col0 = tile * 16;
    if (col0 >= N) return;                        // wave-uniform exit

    const int cN    = col0 + (lane & 15);
    const int c     = (cN < N) ? cN : (N - 1);    // clamp for ragged last tile
    const int khalf = (lane < 16) ? 0 : 16;

    v8f acc = {};
    for (int k0 = 0; k0 < K; k0 += 32) {
        // A: 16x32 f16, row 0 = vec[k0..k0+31], rows 1..15 = 0
        // lanes 0-15 hold K = 0..15 ; lanes 16-31 hold K = 16..31
        v16h a = {};
        if (lane == 0) {
#pragma unroll
            for (int i = 0; i < 8; ++i) {
                a[i]     = (_Float16)vec[k0 + i];
                a[i + 8] = (_Float16)vec[k0 + 8 + i];
            }
        } else if (lane == 16) {
#pragma unroll
            for (int i = 0; i < 8; ++i) {
                a[i]     = (_Float16)vec[k0 + 16 + i];
                a[i + 8] = (_Float16)vec[k0 + 24 + i];
            }
        }
        // B: 32x16 f16 tile; lane = column, half-wave selects K range
        v16h b;
        const int kb = k0 + khalf;
#pragma unroll
        for (int i = 0; i < 16; ++i)
            b[i] = (_Float16)W[(size_t)(kb + i) * N + c];
        if (k0 + 32 < K)
            __builtin_prefetch(&W[(size_t)(kb + 32) * N + c], 0, 1);

        acc = __builtin_amdgcn_wmma_f32_16x16x32_f16(
            /*neg_a=*/false, a, /*neg_b=*/false, b,
            /*c_mod=*/(short)0, acc, /*reuse_a=*/false, /*reuse_b=*/false);
    }

    // D row M=0 lives in acc[0] of lanes 0..15 (N = lane)
    if (lane < 16 && cN < N) {
        float r = acc[0];
        if (bias)  r += bias[cN];
        if (resid) r += resid[cN];
        if (relu)  r = fmaxf(r, 0.0f);
        out[cN] = r;
    }
}

// ---------------------------------------------------------------------------
// Attention for one layer: one block per head, streaming softmax over
// TCACHE cached positions + 1 new position held in the qkv buffer.
// ---------------------------------------------------------------------------
__global__ void attn_kernel(const float* __restrict__ qkv,   // [1536] q,kn,vn
                            const float* __restrict__ kc,    // [T, D]
                            const float* __restrict__ vc,    // [T, D]
                            float* __restrict__ o)           // [512]
{
    const int h   = blockIdx.x;     // 16 heads
    const int tid = threadIdx.x;    // 256 threads

    __shared__ float qsh[DHEAD];
    __shared__ float red[256];
    __shared__ float pbuf[256 * DHEAD];

    if (tid < DHEAD) qsh[tid] = qkv[h * DHEAD + tid] * 0.1767766953f; // 1/sqrt(32)
    __syncthreads();

    float s[9];
    float mx = -3.0e38f;
    int   nt = 0;
    for (int t = tid; t < TCACHE + 1; t += 256) {
        const float* kr = (t < TCACHE) ? (kc + (size_t)t * DMODEL + h * DHEAD)
                                       : (qkv + DMODEL + h * DHEAD);
        float acc = 0.0f;
#pragma unroll
        for (int d = 0; d < DHEAD; d += 4) {
            const float4 k4 = *(const float4*)(kr + d);
            acc += qsh[d] * k4.x + qsh[d + 1] * k4.y + qsh[d + 2] * k4.z + qsh[d + 3] * k4.w;
        }
        s[nt++] = acc;
        mx = fmaxf(mx, acc);
    }

    red[tid] = mx; __syncthreads();
    for (int st = 128; st > 0; st >>= 1) {
        if (tid < st) red[tid] = fmaxf(red[tid], red[tid + st]);
        __syncthreads();
    }
    mx = red[0]; __syncthreads();

    float po[DHEAD];
#pragma unroll
    for (int d = 0; d < DHEAD; ++d) po[d] = 0.0f;
    float ls = 0.0f;
    nt = 0;
    for (int t = tid; t < TCACHE + 1; t += 256) {
        const float p = __expf(s[nt++] - mx);
        ls += p;
        const float* vr = (t < TCACHE) ? (vc + (size_t)t * DMODEL + h * DHEAD)
                                       : (qkv + 2 * DMODEL + h * DHEAD);
#pragma unroll
        for (int d = 0; d < DHEAD; d += 4) {
            const float4 v4 = *(const float4*)(vr + d);
            po[d]     += p * v4.x;
            po[d + 1] += p * v4.y;
            po[d + 2] += p * v4.z;
            po[d + 3] += p * v4.w;
        }
    }

    red[tid] = ls; __syncthreads();
    for (int st = 128; st > 0; st >>= 1) {
        if (tid < st) red[tid] += red[tid + st];
        __syncthreads();
    }
    const float denom = red[0];

#pragma unroll
    for (int d = 0; d < DHEAD; ++d) pbuf[tid * DHEAD + d] = po[d];
    __syncthreads();
    for (int st = 128; st > 0; st >>= 1) {
        if (tid < st) {
#pragma unroll
            for (int d = 0; d < DHEAD; ++d)
                pbuf[tid * DHEAD + d] += pbuf[(tid + st) * DHEAD + d];
        }
        __syncthreads();
    }
    if (tid < DHEAD) o[h * DHEAD + tid] = pbuf[tid] / denom;
}

// ---------------------------------------------------------------------------
// LayerNorm over 512 elements (one block, 512 threads)
// ---------------------------------------------------------------------------
__global__ void layernorm_kernel(const float* __restrict__ in,
                                 const float* __restrict__ g,
                                 const float* __restrict__ b,
                                 float* __restrict__ outx)
{
    __shared__ float red[DMODEL];
    const int tid = threadIdx.x;
    const float v = in[tid];

    red[tid] = v; __syncthreads();
    for (int st = 256; st > 0; st >>= 1) {
        if (tid < st) red[tid] += red[tid + st];
        __syncthreads();
    }
    const float mu = red[0] * (1.0f / DMODEL);
    __syncthreads();

    const float dv = v - mu;
    red[tid] = dv * dv; __syncthreads();
    for (int st = 256; st > 0; st >>= 1) {
        if (tid < st) red[tid] += red[tid + st];
        __syncthreads();
    }
    const float var = red[0] * (1.0f / DMODEL);
    outx[tid] = dv * __frsqrt_rn(var + 1e-5f) * g[tid] + b[tid];
}

// ---------------------------------------------------------------------------
// copy new K/V rows into output
// ---------------------------------------------------------------------------
__global__ void copy_kv_kernel(const float* __restrict__ qkv,
                               float* __restrict__ ko,
                               float* __restrict__ vo)
{
    const int i = threadIdx.x;      // 512
    ko[i] = qkv[DMODEL + i];
    vo[i] = qkv[2 * DMODEL + i];
}

// ---------------------------------------------------------------------------
// rep-penalty + top-k mask + softmax + sample (one block, 1024 threads)
// ---------------------------------------------------------------------------
__device__ __forceinline__ unsigned pcg_hash(unsigned v)
{
    unsigned s = v * 747796405u + 2891336453u;
    s = ((s >> ((s >> 28u) + 4u)) ^ s) * 277803737u;
    return (s >> 22u) ^ s;
}

__device__ __forceinline__ float hash_normal(unsigned i)
{
    const unsigned a = pcg_hash(i * 2u + 1u);
    const unsigned b = pcg_hash(i * 2u + 2u);
    const float u1 = ((float)(a >> 8) + 1.0f) * (1.0f / 16777216.0f);
    const float u2 = (float)(b >> 8) * (1.0f / 16777216.0f);
    return sqrtf(-2.0f * __logf(u1)) * __cosf(6.2831853f * u2);
}

__global__ void sample_kernel(const float* __restrict__ logits,
                              const int* __restrict__ y,
                              float* __restrict__ probs,
                              float* __restrict__ idx_out)
{
    __shared__ float lp[VOCAB + 31];
    __shared__ float red[1024];
    __shared__ int   redi[1024];
    const int tid = threadIdx.x;    // 1024

    for (int i = tid; i < VOCAB; i += 1024) lp[i] = logits[i];
    __syncthreads();

    // repetition penalty: value computed from ORIGINAL logits (dup-safe)
    if (tid < TYLEN) {
        const int tok = y[tid];
        const float v = logits[tok];
        lp[tok] = (v < 0.0f) ? v * REPP : v / REPP;
    }
    __syncthreads();

    // keep candidate c iff fewer than TOPK elements are strictly greater
    float cv[2]; int ck[2]; int cc[2]; int nc = 0;
    for (int c = tid; c < VOCAB; c += 1024) {
        const float vc = lp[c];
        int cnt = 0;
        for (int j = 0; j < VOCAB; ++j) cnt += (lp[j] > vc) ? 1 : 0;
        cv[nc] = vc; ck[nc] = (cnt < TOPK) ? 1 : 0; cc[nc] = c; ++nc;
    }

    float bmax = -3.0e38f;
    for (int i = 0; i < nc; ++i) if (ck[i]) bmax = fmaxf(bmax, cv[i]);
    red[tid] = bmax; __syncthreads();
    for (int st = 512; st > 0; st >>= 1) {
        if (tid < st) red[tid] = fmaxf(red[tid], red[tid + st]);
        __syncthreads();
    }
    const float mx = red[0]; __syncthreads();

    float pv[2]; float psum = 0.0f;
    for (int i = 0; i < nc; ++i) {
        pv[i] = ck[i] ? __expf(cv[i] - mx) : 0.0f;
        psum += pv[i];
    }
    red[tid] = psum; __syncthreads();
    for (int st = 512; st > 0; st >>= 1) {
        if (tid < st) red[tid] += red[tid + st];
        __syncthreads();
    }
    const float pinv = 1.0f / red[0]; __syncthreads();

    float bestr = -3.0e38f; int besti = 0;
    for (int i = 0; i < nc; ++i) {
        const float p = pv[i] * pinv;
        probs[cc[i]] = p;
        const float r = p / hash_normal((unsigned)cc[i]);
        if (r > bestr) { bestr = r; besti = cc[i]; }
    }
    red[tid] = bestr; redi[tid] = besti; __syncthreads();
    for (int st = 512; st > 0; st >>= 1) {
        if (tid < st && red[tid + st] > red[tid]) {
            red[tid] = red[tid + st]; redi[tid] = redi[tid + st];
        }
        __syncthreads();
    }
    if (tid == 0) idx_out[0] = (float)redi[0];
}

// ---------------------------------------------------------------------------
// host driver
// ---------------------------------------------------------------------------
static inline int gemv_blocks(int N)
{
    const int waves = (N + 15) / 16;
    return (waves + 7) / 8;          // 8 waves (256 threads) per block
}

extern "C" void kernel_launch(void* const* d_in, const int* in_sizes, int n_in,
                              void* d_out, int out_size, void* d_ws, size_t ws_size,
                              hipStream_t stream)
{
    (void)in_sizes; (void)n_in; (void)out_size; (void)ws_size;

    const int*   y       = (const int*)  d_in[0];
    const float* k_cache = (const float*)d_in[1];
    const float* v_cache = (const float*)d_in[2];
    /* d_in[3] = y_emb (unused: only the freshly embedded token feeds x) */
    const float* tok_emb = (const float*)d_in[4];
    const float* alpha   = (const float*)d_in[5];
    const float* qkv_w   = (const float*)d_in[6];
    const float* qkv_b   = (const float*)d_in[7];
    const float* out_w   = (const float*)d_in[8];
    const float* out_b   = (const float*)d_in[9];
    const float* ln1_g   = (const float*)d_in[10];
    const float* ln1_b   = (const float*)d_in[11];
    const float* ff1_w   = (const float*)d_in[12];
    const float* ff1_b   = (const float*)d_in[13];
    const float* ff2_w   = (const float*)d_in[14];
    const float* ff2_b   = (const float*)d_in[15];
    const float* ln2_g   = (const float*)d_in[16];
    const float* ln2_b   = (const float*)d_in[17];
    const float* pred_w  = (const float*)d_in[18];

    float* out       = (float*)d_out;
    float* probs_out = out;                        // [1025]
    float* idx_out   = out + VOCAB;                // [1]
    float* knew_out  = out + VOCAB + 1;            // [24*512]
    float* vnew_out  = knew_out + LAYERS * DMODEL; // [24*512]

    float* ws     = (float*)d_ws;
    float* x      = ws;            // 512
    float* qkv    = ws + 512;      // 1536
    float* ovec   = ws + 2048;     // 512
    float* rvec   = ws + 2560;     // 512
    float* hff    = ws + 3072;     // 2048
    float* logits = ws + 5120;     // 1025 (+pad)

    prep_x_kernel<<<1, DMODEL, 0, stream>>>(y, tok_emb, alpha, x);

    for (int l = 0; l < LAYERS; ++l) {
        const size_t qw_off = (size_t)l * DMODEL * 3 * DMODEL;
        const size_t ow_off = (size_t)l * DMODEL * DMODEL;
        const size_t f1_off = (size_t)l * DMODEL * FFDIM;
        const size_t f2_off = (size_t)l * FFDIM * DMODEL;
        const size_t kv_off = (size_t)l * TCACHE * DMODEL;

        gemv_wmma_kernel<<<gemv_blocks(3 * DMODEL), 256, 0, stream>>>(
            x, qkv_w + qw_off, qkv_b + l * 3 * DMODEL, nullptr, qkv,
            DMODEL, 3 * DMODEL, 0);

        copy_kv_kernel<<<1, DMODEL, 0, stream>>>(
            qkv, knew_out + l * DMODEL, vnew_out + l * DMODEL);

        attn_kernel<<<NHEAD, 256, 0, stream>>>(
            qkv, k_cache + kv_off, v_cache + kv_off, ovec);

        gemv_wmma_kernel<<<gemv_blocks(DMODEL), 256, 0, stream>>>(
            ovec, out_w + ow_off, out_b + l * DMODEL, x, rvec,
            DMODEL, DMODEL, 0);
        layernorm_kernel<<<1, DMODEL, 0, stream>>>(
            rvec, ln1_g + l * DMODEL, ln1_b + l * DMODEL, x);

        gemv_wmma_kernel<<<gemv_blocks(FFDIM), 256, 0, stream>>>(
            x, ff1_w + f1_off, ff1_b + l * FFDIM, nullptr, hff,
            DMODEL, FFDIM, 1);
        gemv_wmma_kernel<<<gemv_blocks(DMODEL), 256, 0, stream>>>(
            hff, ff2_w + f2_off, ff2_b + l * DMODEL, x, rvec,
            FFDIM, DMODEL, 0);
        layernorm_kernel<<<1, DMODEL, 0, stream>>>(
            rvec, ln2_g + l * DMODEL, ln2_b + l * DMODEL, x);
    }

    gemv_wmma_kernel<<<gemv_blocks(VOCAB), 256, 0, stream>>>(
        x, pred_w, nullptr, nullptr, logits, DMODEL, VOCAB, 0);

    sample_kernel<<<1, 1024, 0, stream>>>(logits, y, probs_out, idx_out);
}
